// ContrastAwareLocalAttention_59794534695446
// MI455X (gfx1250) — matrix operand order repaired and checked by reading, not compiled, not run on hardware
//
#include <hip/hip_runtime.h>
#include <math.h>

// ---------------- problem constants ----------------
#define BATCH 4
#define HH    96
#define WWID  96
#define HWPIX (HH*WWID)           // 9216
#define NPIXT (BATCH*HWPIX)       // 36864
#define CIN   64
#define CDIM  128
#define NHEAD 8
#define HDIM  16
#define KW    3
#define KK2   9
#define WSZ   7
#define NWINS 196                 // 14*14 windows
#define WTOK  49
#define TOKT  (BATCH*NWINS*WTOK)  // 38416
#define AOUT  648                 // K^4 * NH
#define AOUTP 704                 // AOUT padded to multiple of 64 (N-tile)
#define EPSV  1e-5f
#define SCALEV 0.25f              // HD^-0.5

typedef _Float16 h16;
typedef __attribute__((ext_vector_type(16))) _Float16 v16h;
typedef __attribute__((ext_vector_type(8)))  _Float16 v8h;
typedef __attribute__((ext_vector_type(8)))  float    v8f;

// ---------------- WMMA operand loaders ----------------
// A-matrix 16x32 f16 striping (ISA 7.12.2): lane<16 holds K {0..7, 16..23},
// lane>=16 holds K {8..15, 24..31}.  p already includes the +8 half offset.
__device__ __forceinline__ v16h ldA(const h16* p) {
  v8h lo = *(const v8h*)(p);
  v8h hi = *(const v8h*)(p + 16);
  v16h r;
#pragma unroll
  for (int i = 0; i < 8; ++i) { r[i] = lo[i]; r[8 + i] = hi[i]; }
  return r;
}
// B-matrix 32x16: lane<16 holds K {0..7, 16..23}, lane>=16 holds K {8..15, 24..31}.
// p already includes the +8 half offset for high lanes.
__device__ __forceinline__ v16h ldB(const h16* p) {
  v8h lo = *(const v8h*)(p);
  v8h hi = *(const v8h*)(p + 16);
  v16h r;
#pragma unroll
  for (int i = 0; i < 8; ++i) { r[i] = lo[i]; r[8 + i] = hi[i]; }
  return r;
}

// ---------------- generic WMMA GEMM: C[M,N] = A[M,K] * Bw[N,K]^T (+bias) ----------------
// A, Bw row-major f16; K multiple of 32; M multiple of 16.
// One wave computes a 16x64 strip (4 accumulators, A reused 4x, no in-loop masking:
// Bw must be padded with zero rows up to ceil(N/64)*64).
__global__ void k_wmma_gemm(const h16* __restrict__ A, const h16* __restrict__ Bw,
                            const float* __restrict__ bias,
                            float* __restrict__ C32, h16* __restrict__ C16,
                            int M, int N, int Kd) {
  int lane = threadIdx.x & 31;
  int wid  = threadIdx.x >> 5;
  int mt = blockIdx.x * 4 + wid;
  if (mt * 16 >= M) return;
  int nbase = blockIdx.y * 64;
  int r = lane & 15;
  int hiA = (lane >= 16) ? 8 : 0;
  int hiB = (lane >= 16) ? 8 : 0;
  const h16* arow = A + (size_t)(mt * 16 + r) * Kd + hiA;
  const h16* brow = Bw + (size_t)(nbase + r) * Kd + hiB;
  size_t s16 = (size_t)16 * Kd;
  v8f acc0 = {}, acc1 = {}, acc2 = {}, acc3 = {};
  for (int kk = 0; kk < Kd; kk += 32) {
    __builtin_prefetch((const void*)(arow + kk + 256), 0, 3);
    v16h a  = ldA(arow + kk);
    v16h b0 = ldB(brow + kk);
    v16h b1 = ldB(brow + s16 + kk);
    v16h b2 = ldB(brow + 2 * s16 + kk);
    v16h b3 = ldB(brow + 3 * s16 + kk);
    acc0 = __builtin_amdgcn_wmma_f32_16x16x32_f16(false, a, false, b0, (short)0, acc0, false, false);
    acc1 = __builtin_amdgcn_wmma_f32_16x16x32_f16(false, a, false, b1, (short)0, acc1, false, false);
    acc2 = __builtin_amdgcn_wmma_f32_16x16x32_f16(false, a, false, b2, (short)0, acc2, false, false);
    acc3 = __builtin_amdgcn_wmma_f32_16x16x32_f16(false, a, false, b3, (short)0, acc3, false, false);
  }
  int rbase = mt * 16 + ((lane >= 16) ? 8 : 0);
#pragma unroll
  for (int j = 0; j < 4; ++j) {
    v8f acc = (j == 0) ? acc0 : (j == 1) ? acc1 : (j == 2) ? acc2 : acc3;
    int n = nbase + j * 16 + r;
    if (n >= N) continue;
    float bv = bias ? bias[n] : 0.f;
#pragma unroll
    for (int i = 0; i < 8; ++i) {
      float v = acc[i] + bv;
      size_t idx = (size_t)(rbase + i) * N + n;
      if (C32) C32[idx] = v;
      if (C16) C16[idx] = (h16)v;
    }
  }
}

// ---------------- WMMA 3x3 conv (pad=1), NHWC f16 in, f32 out ----------------
// Weights pre-swizzled to [tap(9)][Cout][Cin]; Cout multiple of 64.
// Out-of-image taps read from a zeroed scratch page (no register masking).
__global__ void k_wmma_conv3(const h16* __restrict__ In, const h16* __restrict__ Wt,
                             const h16* __restrict__ zpage,
                             float* __restrict__ Out, int Cin, int Cout) {
  int lane = threadIdx.x & 31;
  int wid  = threadIdx.x >> 5;
  int mt = blockIdx.x * 4 + wid;
  if (mt * 16 >= NPIXT) return;
  int nbase = blockIdx.y * 64;
  int r = lane & 15;
  int hiA = (lane >= 16) ? 8 : 0;
  int hiB = (lane >= 16) ? 8 : 0;
  int m = mt * 16 + r;
  int b = m / HWPIX;
  int rem = m % HWPIX;
  int y = rem / WWID;
  int x = rem % WWID;
  size_t s16 = (size_t)16 * Cin;
  v8f acc0 = {}, acc1 = {}, acc2 = {}, acc3 = {};
  for (int tap = 0; tap < 9; ++tap) {
    int ys = y + tap / 3 - 1;
    int xs = x + tap % 3 - 1;
    bool ok = (ys >= 0 && ys < HH && xs >= 0 && xs < WWID);
    const h16* arow = ok ? (In + ((size_t)(b * HH + ys) * WWID + xs) * Cin + hiA)
                         : (zpage + hiA);
    const h16* brow = Wt + ((size_t)tap * Cout + nbase + r) * Cin + hiB;
    for (int kk = 0; kk < Cin; kk += 32) {
      v16h a  = ldA(arow + kk);
      v16h b0 = ldB(brow + kk);
      v16h b1 = ldB(brow + s16 + kk);
      v16h b2 = ldB(brow + 2 * s16 + kk);
      v16h b3 = ldB(brow + 3 * s16 + kk);
      acc0 = __builtin_amdgcn_wmma_f32_16x16x32_f16(false, a, false, b0, (short)0, acc0, false, false);
      acc1 = __builtin_amdgcn_wmma_f32_16x16x32_f16(false, a, false, b1, (short)0, acc1, false, false);
      acc2 = __builtin_amdgcn_wmma_f32_16x16x32_f16(false, a, false, b2, (short)0, acc2, false, false);
      acc3 = __builtin_amdgcn_wmma_f32_16x16x32_f16(false, a, false, b3, (short)0, acc3, false, false);
    }
  }
  int rbase = mt * 16 + ((lane >= 16) ? 8 : 0);
#pragma unroll
  for (int j = 0; j < 4; ++j) {
    v8f acc = (j == 0) ? acc0 : (j == 1) ? acc1 : (j == 2) ? acc2 : acc3;
    int n = nbase + j * 16 + r;
#pragma unroll
    for (int i = 0; i < 8; ++i)
      Out[(size_t)(rbase + i) * Cout + n] = acc[i];
  }
}

// ---------------- BatchNorm (train mode) ----------------
__global__ void k_bn_stats(const float* __restrict__ in, int Np, int C,
                           float* __restrict__ mean, float* __restrict__ istd) {
  int c = blockIdx.x;
  __shared__ float s1[256], s2[256];
  float a = 0.f, b = 0.f;
  for (int i = threadIdx.x; i < Np; i += 256) {
    float v = in[(size_t)i * C + c];
    a += v; b += v * v;
  }
  s1[threadIdx.x] = a; s2[threadIdx.x] = b;
  __syncthreads();
  for (int st = 128; st > 0; st >>= 1) {
    if (threadIdx.x < st) { s1[threadIdx.x] += s1[threadIdx.x + st]; s2[threadIdx.x] += s2[threadIdx.x + st]; }
    __syncthreads();
  }
  if (threadIdx.x == 0) {
    float mm = s1[0] / Np;
    float var = s2[0] / Np - mm * mm;
    mean[c] = mm;
    istd[c] = rsqrtf(var + EPSV);
  }
}

// act: 0=none, 1=relu, 2=exact gelu.  Optional f16 out (strided, channel offset) + f32 out.
__global__ void k_bn_apply(const float* __restrict__ in, int Np, int C,
                           const float* __restrict__ mean, const float* __restrict__ istd,
                           const float* __restrict__ g, const float* __restrict__ bb, int act,
                           h16* __restrict__ out16, int ostride, int ooff,
                           float* __restrict__ out32) {
  size_t t = (size_t)blockIdx.x * blockDim.x + threadIdx.x;
  if (t >= (size_t)Np * C) return;
  int c = (int)(t % C);
  size_t i = t / C;
  float v = (in[t] - mean[c]) * istd[c] * g[c] + bb[c];
  if (act == 1) v = fmaxf(v, 0.f);
  else if (act == 2) v = 0.5f * v * (1.f + erff(v * 0.70710678f));
  if (out16) out16[i * ostride + ooff + c] = (h16)v;
  if (out32) out32[t] = v;
}

// ---------------- LayerNorm over channels (per token) ----------------
__global__ void k_layernorm(const h16* __restrict__ in, const float* __restrict__ g,
                            const float* __restrict__ bb, h16* __restrict__ out, int C) {
  int t = blockIdx.x * blockDim.x + threadIdx.x;
  if (t >= NPIXT) return;
  const h16* p = in + (size_t)t * C;
  float s = 0.f, ss = 0.f;
  for (int c = 0; c < C; ++c) { float v = (float)p[c]; s += v; ss += v * v; }
  float m = s / C;
  float inv = rsqrtf(ss / C - m * m + EPSV);
  h16* o = out + (size_t)t * C;
  for (int c = 0; c < C; ++c)
    o[c] = (h16)(((float)p[c] - m) * inv * g[c] + bb[c]);
}

// ---------------- window gather / scatter (7x7 windows over padded 98x98) ----------------
__device__ __forceinline__ void tok2pix(int tok, int& b, int& y, int& x) {
  b = tok / (NWINS * WTOK);
  int r = tok % (NWINS * WTOK);
  int win = r / WTOK, tt = r % WTOK;
  y = (win / 14) * WSZ + tt / WSZ;
  x = (win % 14) * WSZ + tt % WSZ;
}

__global__ void k_win_gather(const h16* __restrict__ xf, h16* __restrict__ xw) {
  size_t t = (size_t)blockIdx.x * blockDim.x + threadIdx.x;
  if (t >= (size_t)TOKT * CDIM) return;
  int c = (int)(t % CDIM);
  int tok = (int)(t / CDIM);
  int b, y, x; tok2pix(tok, b, y, x);
  xw[t] = (y < HH && x < WWID) ? xf[((size_t)(b * HH + y) * WWID + x) * CDIM + c] : (h16)0.f;
}

__global__ void k_win_scatter(const h16* __restrict__ sp, float* __restrict__ xh) {
  size_t t = (size_t)blockIdx.x * blockDim.x + threadIdx.x;
  if (t >= (size_t)TOKT * CDIM) return;
  int c = (int)(t % CDIM);
  int tok = (int)(t / CDIM);
  int b, y, x; tok2pix(tok, b, y, x);
  if (y < HH && x < WWID)
    xh[((size_t)(b * HH + y) * WWID + x) * CDIM + c] = (float)sp[t];
}

// ---------------- window self-attention: one block per (b, win, head) ----------------
__global__ void k_win_attn(const h16* __restrict__ q, const h16* __restrict__ k,
                           const h16* __restrict__ v, h16* __restrict__ o) {
  int id = blockIdx.x;
  int h = id % NHEAD;
  int t = id / NHEAD;
  int win = t % NWINS;
  int b = t / NWINS;
  __shared__ float ksh[WTOK][HDIM];
  __shared__ float vsh[WTOK][HDIM];
  __shared__ float ssh[WTOK][WTOK];
  int base = (b * NWINS + win) * WTOK;
  for (int i = threadIdx.x; i < WTOK * HDIM; i += 64) {
    int row = i / HDIM, col = i % HDIM;
    ksh[row][col] = (float)k[(size_t)(base + row) * CDIM + h * HDIM + col];
    vsh[row][col] = (float)v[(size_t)(base + row) * CDIM + h * HDIM + col];
  }
  __syncthreads();
  int tq = threadIdx.x;
  if (tq < WTOK) {
    float qr[HDIM];
#pragma unroll
    for (int e = 0; e < HDIM; ++e) qr[e] = (float)q[(size_t)(base + tq) * CDIM + h * HDIM + e];
    float mx = -3.0e38f;
    for (int j = 0; j < WTOK; ++j) {
      float s = 0.f;
#pragma unroll
      for (int e = 0; e < HDIM; ++e) s += qr[e] * ksh[j][e];
      s *= SCALEV;
      ssh[tq][j] = s;
      mx = fmaxf(mx, s);
    }
    float sum = 0.f;
    for (int j = 0; j < WTOK; ++j) { float p = __expf(ssh[tq][j] - mx); ssh[tq][j] = p; sum += p; }
    float inv = 1.f / sum;
    float orow[HDIM];
#pragma unroll
    for (int e = 0; e < HDIM; ++e) orow[e] = 0.f;
    for (int j = 0; j < WTOK; ++j) {
      float p = ssh[tq][j] * inv;
#pragma unroll
      for (int e = 0; e < HDIM; ++e) orow[e] += p * vsh[j][e];
    }
#pragma unroll
    for (int e = 0; e < HDIM; ++e)
      o[(size_t)(base + tq) * CDIM + h * HDIM + e] = (h16)orow[e];
  }
}

// ---------------- fused involution: int-linear + softmax + a@v_unf + fold(atomic) ----------------
__global__ void k_involution(const h16* __restrict__ afg, const h16* __restrict__ abg,
                             const float* __restrict__ intw, const float* __restrict__ intb,
                             const h16* __restrict__ vv, float* __restrict__ folded) {
  int pix = blockIdx.x;
  int b = pix / HWPIX;
  int rem = pix % HWPIX;
  int y = rem / WWID, x = rem % WWID;
  __shared__ float ash[AOUT];
  __shared__ float wsh[136];
  int tid = threadIdx.x;
  for (int i = tid; i < 136; i += 64) wsh[i] = (i < 128) ? intw[i] : intb[i - 128];
  __syncthreads();
  const h16* fa = afg + (size_t)pix * AOUT;
  const h16* ba = abg + (size_t)pix * AOUT;
  for (int idx = tid; idx < AOUT; idx += 64) {
    int hp = idx / 81, r = idx % 81;
    float s = wsh[128 + hp];
#pragma unroll
    for (int c = 0; c < 8; ++c) s += wsh[hp * 16 + c] * (SCALEV * (float)fa[c * 81 + r]);
#pragma unroll
    for (int c = 0; c < 8; ++c) s += wsh[hp * 16 + 8 + c] * (SCALEV * (float)ba[c * 81 + r]);
    ash[idx] = s;
  }
  __syncthreads();
  for (int row = tid; row < 72; row += 64) {      // softmax over last K2 axis
    float* p = &ash[row * 9];
    float mx = p[0];
#pragma unroll
    for (int j = 1; j < 9; ++j) mx = fmaxf(mx, p[j]);
    float sm = 0.f;
#pragma unroll
    for (int j = 0; j < 9; ++j) { float e = __expf(p[j] - mx); p[j] = e; sm += e; }
    float inv = 1.f / sm;
#pragma unroll
    for (int j = 0; j < 9; ++j) p[j] *= inv;
  }
  __syncthreads();
  int e = tid & 15;
  for (int h = tid >> 4; h < NHEAD; h += 4) {
    float vr[9];
#pragma unroll
    for (int j = 0; j < 9; ++j) {
      int ys = y + j / 3 - 1, xs = x + j % 3 - 1;
      bool ok = (ys >= 0 && ys < HH && xs >= 0 && xs < WWID);
      vr[j] = ok ? (float)vv[((size_t)(b * HH + ys) * WWID + xs) * CDIM + h * HDIM + e] : 0.f;
    }
#pragma unroll
    for (int i2 = 0; i2 < 9; ++i2) {
      float acc = 0.f;
#pragma unroll
      for (int j = 0; j < 9; ++j) acc += ash[h * 81 + i2 * 9 + j] * vr[j];
      int yt = y + i2 / 3 - 1, xt = x + i2 % 3 - 1;
      if (yt >= 0 && yt < HH && xt >= 0 && xt < WWID)
        atomicAdd(&folded[((size_t)(b * HH + yt) * WWID + xt) * CDIM + h * HDIM + e], acc);
    }
  }
}

// ---------------- layout / cast helpers ----------------
__global__ void k_tohwc16(const float* __restrict__ in, h16* __restrict__ out, int C) {
  size_t t = (size_t)blockIdx.x * blockDim.x + threadIdx.x;
  if (t >= (size_t)NPIXT * C) return;
  int c = (int)(t % C);
  size_t pix = t / C;
  int b = (int)(pix / HWPIX);
  int r = (int)(pix % HWPIX);
  out[t] = (h16)in[((size_t)b * C + c) * HWPIX + r];
}

__global__ void k_cast16(const float* __restrict__ in, h16* __restrict__ out, int n) {
  int t = blockIdx.x * blockDim.x + threadIdx.x;
  if (t < n) out[t] = (h16)in[t];
}

// weight cast with zero row padding: out [Np, K], in [N, K]
__global__ void k_castpadw(const float* __restrict__ in, h16* __restrict__ out,
                           int N, int K, int Np) {
  int t = blockIdx.x * blockDim.x + threadIdx.x;
  if (t >= Np * K) return;
  int n = t / K, k = t % K;
  out[t] = (n < N) ? (h16)in[n * K + k] : (h16)0.f;
}

__global__ void k_convw(const float* __restrict__ in, h16* __restrict__ out, int O, int I) {
  int t = blockIdx.x * blockDim.x + threadIdx.x;
  if (t >= O * I * 9) return;
  int tap = t % 9;
  int r = t / 9;
  int i = r % I, o = r / I;
  out[((size_t)tap * O + o) * I + i] = (h16)in[t];
}

__global__ void k_paeff(const float* __restrict__ in, h16* __restrict__ out) {
  int t = blockIdx.x * blockDim.x + threadIdx.x;  // 128*128
  if (t >= CDIM * CDIM) return;
  int n = t / CDIM, k = t % CDIM;
  out[t] = (h16)(in[n * 256 + k] + in[n * 256 + 128 + k]);
}

__global__ void k_zero(float* __restrict__ p, int n) {
  int t = blockIdx.x * blockDim.x + threadIdx.x;
  if (t < n) p[t] = 0.f;
}

__global__ void k_zero16(h16* __restrict__ p, int n) {
  int t = blockIdx.x * blockDim.x + threadIdx.x;
  if (t < n) p[t] = (h16)0.f;
}

// ---------------- final: gelu(bn(o2conv) + xs), NHWC -> NCHW f32 ----------------
__global__ void k_final(const float* __restrict__ conv, const float* __restrict__ mean,
                        const float* __restrict__ istd, const float* __restrict__ g,
                        const float* __restrict__ bb, const float* __restrict__ xs,
                        float* __restrict__ out) {
  size_t t = (size_t)blockIdx.x * blockDim.x + threadIdx.x;
  if (t >= (size_t)BATCH * CDIM * HWPIX) return;
  int b = (int)(t / ((size_t)CDIM * HWPIX));
  int r = (int)(t % ((size_t)CDIM * HWPIX));
  int c = r / HWPIX;
  int pq = r % HWPIX;
  size_t i = (size_t)b * HWPIX + pq;
  float v = (conv[i * CDIM + c] - mean[c]) * istd[c] * g[c] + bb[c] + xs[i * CDIM + c];
  out[t] = 0.5f * v * (1.f + erff(v * 0.70710678f));
}

// =====================================================================
extern "C" void kernel_launch(void* const* d_in, const int* in_sizes, int n_in,
                              void* d_out, int out_size, void* d_ws, size_t ws_size,
                              hipStream_t stream) {
  (void)in_sizes; (void)n_in; (void)out_size; (void)ws_size;
  auto P = [&](int i) { return (const float*)d_in[i]; };
  // input indices (dict order): 0 x, 1 fg, 2 bg, then params in insertion order
  // 3 r1_w 4 r1_g 5 r1_b 6 r2_w 7 r2_g 8 r2_b 9 m1_w 10 m1_g 11 m1_b
  // 12 m2_w 13 m2_g 14 m2_b 15 f_w 16 f_g 17 f_b 18 q_w 19 q_b 20 k_w 21 k_b
  // 22 vv_w 23 vv_b 24 sp_w 25 sp_b 26 bn_g 27 bn_b 28 v_w 29 v_b
  // 30 pa_w 31 pa_g 32 pa_b 33 ln_g 34 ln_b 35 afg_w 36 afg_b 37 abg_w 38 abg_b
  // 39 int_w 40 int_b 41 proj_w 42 proj_b 43 o1_w 44 o1_g 45 o1_b 46 o2_w 47 o2_g 48 o2_b

  char* wsb = (char*)d_ws;
  size_t off = 0;
  auto alloc = [&](size_t bytes) -> void* {
    void* p = wsb + off;
    off += (bytes + 255) & ~(size_t)255;
    return p;
  };

  // weights (f16)
  h16* w_r1   = (h16*)alloc((size_t)64 * 64 * 2);
  h16* w_r2   = (h16*)alloc((size_t)9 * 128 * 64 * 2);
  h16* w_m1   = (h16*)alloc((size_t)9 * 128 * 64 * 2);
  h16* w_m2   = (h16*)alloc((size_t)9 * 128 * 128 * 2);
  h16* w_f    = (h16*)alloc((size_t)128 * 256 * 2);
  h16* w_q    = (h16*)alloc((size_t)128 * 128 * 2);
  h16* w_k    = (h16*)alloc((size_t)128 * 128 * 2);
  h16* w_vv   = (h16*)alloc((size_t)128 * 128 * 2);
  h16* w_sp   = (h16*)alloc((size_t)128 * 128 * 2);
  h16* w_v    = (h16*)alloc((size_t)128 * 128 * 2);
  h16* w_proj = (h16*)alloc((size_t)128 * 128 * 2);
  h16* w_pa   = (h16*)alloc((size_t)128 * 128 * 2);
  h16* w_afg  = (h16*)alloc((size_t)AOUTP * 128 * 2);   // zero-padded rows
  h16* w_abg  = (h16*)alloc((size_t)AOUTP * 128 * 2);   // zero-padded rows
  h16* w_o1   = (h16*)alloc((size_t)9 * 128 * 128 * 2);
  h16* w_o2   = (h16*)alloc((size_t)9 * 128 * 128 * 2);
  h16* zpage  = (h16*)alloc((size_t)256 * 2);           // zero page for conv border taps
  // activations
  h16*   x16     = (h16*)alloc((size_t)NPIXT * 64 * 2);
  float* TMPF32  = (float*)alloc((size_t)NPIXT * 128 * 4);  // shared conv temp + fold acc
  h16*   xs1_16  = (h16*)alloc((size_t)NPIXT * 64 * 2);
  float* xs_f32  = (float*)alloc((size_t)NPIXT * 128 * 4);
  h16*   xm16    = (h16*)alloc((size_t)NPIXT * 128 * 2);
  h16*   xcat16  = (h16*)alloc((size_t)NPIXT * 256 * 2);
  h16*   xf16    = (h16*)alloc((size_t)NPIXT * 128 * 2);
  h16*   xfwin   = (h16*)alloc((size_t)TOKT * 128 * 2);
  h16*   q16     = (h16*)alloc((size_t)TOKT * 128 * 2);
  h16*   k16b    = (h16*)alloc((size_t)TOKT * 128 * 2);
  h16*   v16b    = (h16*)alloc((size_t)TOKT * 128 * 2);
  h16*   o16     = (h16*)alloc((size_t)TOKT * 128 * 2);
  h16*   sp16    = (h16*)alloc((size_t)TOKT * 128 * 2);
  h16*   xh16    = (h16*)alloc((size_t)NPIXT * 128 * 2);
  h16*   vout16  = (h16*)alloc((size_t)NPIXT * 128 * 2);
  h16*   fgbg16  = (h16*)alloc((size_t)NPIXT * 128 * 2);
  h16*   pan16   = (h16*)alloc((size_t)NPIXT * 128 * 2);
  h16*   ln16    = (h16*)alloc((size_t)NPIXT * 128 * 2);
  h16*   afg16   = (h16*)alloc((size_t)NPIXT * AOUT * 2);
  h16*   abg16   = (h16*)alloc((size_t)NPIXT * AOUT * 2);
  h16*   fold16  = (h16*)alloc((size_t)NPIXT * 128 * 2);
  h16*   pj16    = (h16*)alloc((size_t)NPIXT * 128 * 2);
  h16*   o1_16   = (h16*)alloc((size_t)NPIXT * 128 * 2);
  float* ws_mean = (float*)alloc(128 * 4);
  float* ws_istd = (float*)alloc(128 * 4);

  auto cdiv = [](int a, int b) { return (a + b - 1) / b; };
  auto gemm = [&](const h16* A, const h16* Bw, const float* bias, float* C32, h16* C16,
                  int M, int N, int Kd) {
    dim3 g(cdiv(M, 64), cdiv(N, 64));
    k_wmma_gemm<<<g, 128, 0, stream>>>(A, Bw, bias, C32, C16, M, N, Kd);
  };
  auto conv3 = [&](const h16* In, const h16* Wt, float* Out, int Cin, int Cout) {
    dim3 g(NPIXT / 64, Cout / 64);
    k_wmma_conv3<<<g, 128, 0, stream>>>(In, Wt, zpage, Out, Cin, Cout);
  };
  auto stats = [&](const float* in, int C) {
    k_bn_stats<<<C, 256, 0, stream>>>(in, NPIXT, C, ws_mean, ws_istd);
  };
  auto bnapp = [&](const float* in, int C, int gi, int bi, int act, h16* out16,
                   int ostride, int ooff, float* out32) {
    int n = NPIXT * C;
    k_bn_apply<<<cdiv(n, 256), 256, 0, stream>>>(in, NPIXT, C, ws_mean, ws_istd,
                                                 P(gi), P(bi), act, out16, ostride, ooff, out32);
  };

  // ---- weight / input conversion ----
  k_zero16<<<1, 256, 0, stream>>>(zpage, 256);
  k_cast16<<<cdiv(64 * 64, 256), 256, 0, stream>>>(P(3), w_r1, 64 * 64);
  k_convw<<<cdiv(128 * 64 * 9, 256), 256, 0, stream>>>(P(6), w_r2, 128, 64);
  k_convw<<<cdiv(128 * 64 * 9, 256), 256, 0, stream>>>(P(9), w_m1, 128, 64);
  k_convw<<<cdiv(128 * 128 * 9, 256), 256, 0, stream>>>(P(12), w_m2, 128, 128);
  k_cast16<<<cdiv(128 * 256, 256), 256, 0, stream>>>(P(15), w_f, 128 * 256);
  k_cast16<<<cdiv(16384, 256), 256, 0, stream>>>(P(18), w_q, 16384);
  k_cast16<<<cdiv(16384, 256), 256, 0, stream>>>(P(20), w_k, 16384);
  k_cast16<<<cdiv(16384, 256), 256, 0, stream>>>(P(22), w_vv, 16384);
  k_cast16<<<cdiv(16384, 256), 256, 0, stream>>>(P(24), w_sp, 16384);
  k_cast16<<<cdiv(16384, 256), 256, 0, stream>>>(P(28), w_v, 16384);
  k_cast16<<<cdiv(16384, 256), 256, 0, stream>>>(P(41), w_proj, 16384);
  k_paeff<<<cdiv(16384, 256), 256, 0, stream>>>(P(30), w_pa);
  k_castpadw<<<cdiv(AOUTP * 128, 256), 256, 0, stream>>>(P(35), w_afg, AOUT, 128, AOUTP);
  k_castpadw<<<cdiv(AOUTP * 128, 256), 256, 0, stream>>>(P(37), w_abg, AOUT, 128, AOUTP);
  k_convw<<<cdiv(128 * 128 * 9, 256), 256, 0, stream>>>(P(43), w_o1, 128, 128);
  k_convw<<<cdiv(128 * 128 * 9, 256), 256, 0, stream>>>(P(46), w_o2, 128, 128);
  k_tohwc16<<<cdiv(NPIXT * 64, 256), 256, 0, stream>>>(P(0), x16, 64);

  // ---- xs = cbr(cbr(x, r1), r2) ----
  gemm(x16, w_r1, nullptr, TMPF32, nullptr, NPIXT, 64, 64);
  stats(TMPF32, 64);
  bnapp(TMPF32, 64, 4, 5, 0, xs1_16, 64, 0, nullptr);
  conv3(xs1_16, w_r2, TMPF32, 64, 128);
  stats(TMPF32, 128);
  bnapp(TMPF32, 128, 7, 8, 0, xcat16, 256, 128, xs_f32);   // xs -> cat[:,128:], keep f32

  // ---- xm = cbr(cbr(x, m1, relu), m2) ----
  conv3(x16, w_m1, TMPF32, 64, 128);
  stats(TMPF32, 128);
  bnapp(TMPF32, 128, 10, 11, 1, xm16, 128, 0, nullptr);
  conv3(xm16, w_m2, TMPF32, 128, 128);
  stats(TMPF32, 128);
  bnapp(TMPF32, 128, 13, 14, 0, xcat16, 256, 0, nullptr);  // xm -> cat[:,0:128]

  // ---- xf = gelu(bn(conv1x1(cat, f))) ----
  gemm(xcat16, w_f, nullptr, TMPF32, nullptr, NPIXT, 128, 256);
  stats(TMPF32, 128);
  bnapp(TMPF32, 128, 16, 17, 2, xf16, 128, 0, nullptr);

  // ---- local window self-attention ----
  k_win_gather<<<cdiv(TOKT * 128, 256), 256, 0, stream>>>(xf16, xfwin);
  gemm(xfwin, w_q, P(19), nullptr, q16, TOKT, 128, 128);
  gemm(xfwin, w_k, P(21), nullptr, k16b, TOKT, 128, 128);
  gemm(xfwin, w_vv, P(23), nullptr, v16b, TOKT, 128, 128);
  k_win_attn<<<BATCH * NWINS * NHEAD, 64, 0, stream>>>(q16, k16b, v16b, o16);
  gemm(o16, w_sp, P(25), nullptr, sp16, TOKT, 128, 128);
  k_win_scatter<<<cdiv(TOKT * 128, 256), 256, 0, stream>>>(sp16, TMPF32);  // xh (pre-bn)
  stats(TMPF32, 128);
  bnapp(TMPF32, 128, 26, 27, 0, xh16, 128, 0, nullptr);
  gemm(xh16, w_v, P(29), nullptr, vout16, NPIXT, 128, 128);  // v values

  // ---- compute_attn(fg) / compute_attn(bg) ----
  k_tohwc16<<<cdiv(NPIXT * 128, 256), 256, 0, stream>>>(P(1), fgbg16, 128);
  gemm(fgbg16, w_pa, nullptr, TMPF32, nullptr, NPIXT, 128, 128);
  stats(TMPF32, 128);
  bnapp(TMPF32, 128, 31, 32, 1, pan16, 128, 0, nullptr);
  k_layernorm<<<cdiv(NPIXT, 256), 256, 0, stream>>>(pan16, P(33), P(34), ln16, 128);
  gemm(ln16, w_afg, P(36), nullptr, afg16, NPIXT, AOUT, 128);

  k_tohwc16<<<cdiv(NPIXT * 128, 256), 256, 0, stream>>>(P(2), fgbg16, 128);
  gemm(fgbg16, w_pa, nullptr, TMPF32, nullptr, NPIXT, 128, 128);
  stats(TMPF32, 128);
  bnapp(TMPF32, 128, 31, 32, 1, pan16, 128, 0, nullptr);
  k_layernorm<<<cdiv(NPIXT, 256), 256, 0, stream>>>(pan16, P(33), P(34), ln16, 128);
  gemm(ln16, w_abg, P(38), nullptr, abg16, NPIXT, AOUT, 128);

  // ---- involution: int-linear + softmax + a@v_unf + fold (atomic scatter) ----
  k_zero<<<cdiv(NPIXT * 128, 256), 256, 0, stream>>>(TMPF32, NPIXT * 128);
  k_involution<<<NPIXT, 64, 0, stream>>>(afg16, abg16, P(39), P(40), vout16, TMPF32);
  k_cast16<<<cdiv(NPIXT * 128, 256), 256, 0, stream>>>(TMPF32, fold16, NPIXT * 128);

  // ---- proj + output convs + residual gelu ----
  gemm(fold16, w_proj, P(42), nullptr, pj16, NPIXT, 128, 128);
  conv3(pj16, w_o1, TMPF32, 128, 128);
  stats(TMPF32, 128);
  bnapp(TMPF32, 128, 44, 45, 1, o1_16, 128, 0, nullptr);
  conv3(o1_16, w_o2, TMPF32, 128, 128);
  stats(TMPF32, 128);
  k_final<<<cdiv(NPIXT * 128, 256), 256, 0, stream>>>(TMPF32, ws_mean, ws_istd,
                                                      P(47), P(48), xs_f32, (float*)d_out);
}
